// Seq2SeqRNN_68590627717751
// MI455X (gfx1250) — matrix operand order repaired and checked by reading, not compiled, not run on hardware
//
#include <hip/hip_runtime.h>
#include <cstdint>
#include <cstddef>

// ---------------------------------------------------------------------------
// Seq2Seq GRU encoder/decoder with attention for MI455X (gfx1250).
//
//  * All GEMMs -> v_wmma_f32_16x16x32_bf16 (wave32, 16x16 tiles).
//  * GEMM v2: LDS double-buffered 64x64 tile staging. Tiles are moved by the
//    Tensor Data Mover (tensor_load_to_lds, TENSORcnt) when the builtin is
//    available, so global->LDS runs async to the WMMA stream; fallback is a
//    cooperative copy. TDM pad feature gives a 144B LDS row stride ->
//    bank-conflict-free ds_load_b128 fragment reads.
//  * Attention: W_attn = [W_h | W_enc]; Ue = enc @ W_enc^T hoisted out of the
//    decoder loop (saves ~590 GFLOP vs naive per-step concat energy).
//  * Vocab projection deferred: decoder stores [h2|w|et] rows, then one
//    batched 1984x3584x32000 bf16 GEMM reads W_out (229 MB) exactly once
//    (~10us at 23.3 TB/s) instead of 31x 458 MB fp32 passes.
// ---------------------------------------------------------------------------

typedef unsigned short u16;
typedef __attribute__((ext_vector_type(16))) __bf16 v16bf;
typedef __attribute__((ext_vector_type(8)))  float  v8f;
typedef unsigned int tdm_v4u __attribute__((ext_vector_type(4)));
typedef int          tdm_v8i __attribute__((ext_vector_type(8)));
typedef int          tdm_v4i __attribute__((ext_vector_type(4)));

union BF16x16 { v16bf v; u16 u[16]; uint4 q[2]; };

__device__ __forceinline__ u16 f2bf(float x) {
  unsigned u = __float_as_uint(x);
  u += 0x7FFFu + ((u >> 16) & 1u);       // round-to-nearest-even
  return (u16)(u >> 16);
}
__device__ __forceinline__ float sigmoidf_(float x) { return 1.f / (1.f + __expf(-x)); }

#if defined(__AMDGCN__) && __has_builtin(__builtin_amdgcn_tensor_load_to_lds) && \
    __has_builtin(__builtin_amdgcn_s_wait_tensorcnt)
#define USE_TDM 1
#else
#define USE_TDM 0
#endif

// LDS tile: 64 rows x 64 bf16 (128B data) padded to 144B row stride
// (TDM pad: 4 dwords after every 32 dwords) -> conflict-free b128 reads.
#define TILE_ROW_B   144
#define TILE_ROW_U16 72
#define BK           64

#if USE_TDM
// Issue one TDM 2D tile load: 64 rows x 64 bf16 from row-major (rows x ldK)
// global matrix into LDS at lds_off with padded 144B row stride.
__device__ __forceinline__ void tdm_issue(const u16* gsrc, unsigned lds_off,
                                          unsigned ldK, unsigned rows)
{
  unsigned long long ga = (unsigned long long)(uintptr_t)gsrc;
  tdm_v4u g0;
  g0[0] = 1u;                                   // count=1, load, no gather
  g0[1] = lds_off;                              // lds_addr
  g0[2] = (unsigned)ga;                         // global_addr[31:0]
  g0[3] = ((unsigned)(ga >> 32) & 0x01FFFFFFu)  // global_addr[56:32]
          | (2u << 30);                         // type = 2 (image)
  const unsigned td0 = ldK, td1 = rows;         // tensor dims (OOB guard)
  const unsigned tile0 = 64, tile1 = 64;        // 64 elems x 64 rows
  const unsigned long long str0 = ldK;          // dim0 stride (elements)
  tdm_v8i g1;
  g1[0] = (int)((1u << 16)      // data_size = 2 bytes
          | (1u << 20)          // pad_enable
          | (4u << 22)          // pad_interval = 32 dwords (128B)
          | (3u << 25));        // pad_amount   = 4 dwords (16B)
  g1[1] = (int)((td0 & 0xFFFFu) << 16);               // tensor_dim0[15:0]
  g1[2] = (int)((td0 >> 16) | ((td1 & 0xFFFFu) << 16));
  g1[3] = (int)((td1 >> 16) | (tile0 << 16));         // tile_dim0
  g1[4] = (int)tile1;                                 // tile_dim1, tile_dim2=0
  g1[5] = (int)(unsigned)str0;                        // dim0 stride lo
  g1[6] = (int)((unsigned)(str0 >> 32) & 0xFFFFu);    // stride hi; dim1 stride=0
  g1[7] = 0;
  tdm_v4i z4 = {0, 0, 0, 0};
#if __clang_major__ >= 23
  tdm_v8i z8 = {0, 0, 0, 0, 0, 0, 0, 0};
  __builtin_amdgcn_tensor_load_to_lds(g0, g1, z4, z4, z8, 0);
#else
  __builtin_amdgcn_tensor_load_to_lds(g0, g1, z4, z4, 0);
#endif
}
#endif

// Cooperative fallback: same tile + 144B padded layout via ds_store_b128.
__device__ __forceinline__ void fill_tile_sync(u16* dst, const u16* src,
                                               int ldK, int tid)
{
  for (int i = tid; i < 512; i += 128) {          // 64 rows x 8 uint4
    int row = i >> 3, c = i & 7;
    *(uint4*)((char*)dst + row * TILE_ROW_B + c * 16) =
        *(const uint4*)((const char*)src + (size_t)row * ldK * 2 + c * 16);
  }
}

// ---------------------------------------------------------------------------
// WMMA GEMM:  C[M,N] = act(A[M,K] * W[N,K]^T + bias[N])
// A, W bf16 row-major. Block = 128 threads = 4 waves; block tile 64x64;
// wave w owns rows [m0+16w, m0+16w+16). K staged through LDS in 64-wide
// double-buffered tiles. Requires M%64==0, N%64==0, K%64==0 (all call sites).
// ---------------------------------------------------------------------------
__global__ __launch_bounds__(128)
void k_gemm(const u16* __restrict__ A, const u16* __restrict__ W,
            const float* __restrict__ bias, float* __restrict__ C,
            u16* __restrict__ Cbf, int M, int N, int K, int act)
{
  __shared__ u16 ldsA[2][64 * TILE_ROW_U16];
  __shared__ u16 ldsB[2][64 * TILE_ROW_U16];

  const int n0   = blockIdx.x * 64;
  const int m0   = blockIdx.y * 64;
  const int tid  = threadIdx.x;
  const int wave = tid >> 5;
  const int lane = tid & 31;
  const int half = lane >> 4;          // 0: lanes 0-15, 1: lanes 16-31
  const int l16  = lane & 15;

  v8f acc[4];
  const v8f vzero = {0.f,0.f,0.f,0.f,0.f,0.f,0.f,0.f};
  for (int t = 0; t < 4; ++t) acc[t] = vzero;

  const u16* Abase = A + (size_t)m0 * K;
  const u16* Wbase = W + (size_t)n0 * K;
  const int nstages = K / BK;

  // ---- prologue: stage 0 -> buffer 0 ----
#if USE_TDM
  if (tid < 32) {
    tdm_issue(Abase, (unsigned)(uintptr_t)&ldsA[0][0], (unsigned)K, (unsigned)M);
    tdm_issue(Wbase, (unsigned)(uintptr_t)&ldsB[0][0], (unsigned)K, (unsigned)N);
    __builtin_amdgcn_s_wait_tensorcnt(0);
  }
#else
  fill_tile_sync(ldsA[0], Abase, K, tid);
  fill_tile_sync(ldsB[0], Wbase, K, tid);
#endif
  __syncthreads();

  for (int s = 0; s < nstages; ++s) {
    const int cur = s & 1;

    // ---- async prefetch of stage s+1 into the other buffer ----
    if (s + 1 < nstages) {
#if USE_TDM
      if (tid < 32) {
        tdm_issue(Abase + (size_t)(s + 1) * BK,
                  (unsigned)(uintptr_t)&ldsA[cur ^ 1][0], (unsigned)K, (unsigned)M);
        tdm_issue(Wbase + (size_t)(s + 1) * BK,
                  (unsigned)(uintptr_t)&ldsB[cur ^ 1][0], (unsigned)K, (unsigned)N);
      }
#else
      fill_tile_sync(ldsA[cur ^ 1], Abase + (size_t)(s + 1) * BK, K, tid);
      fill_tile_sync(ldsB[cur ^ 1], Wbase + (size_t)(s + 1) * BK, K, tid);
#endif
    }

    // ---- compute stage s from LDS: 2 k-substeps x 4 n-tiles ----
    const char* Arow = (const char*)&ldsA[cur][0] + (wave * 16 + l16) * TILE_ROW_B;
    const char* Bt   = (const char*)&ldsB[cur][0];
    for (int kk = 0; kk < 2; ++kk) {
      // A fragment (16-bit 16x32 layout, ISA 7.12.2):
      //   half=0 -> k = 0..7,16..23 ; half=1 -> k = 8..15,24..31 (bytes below)
      BF16x16 a;
      a.q[0] = *(const uint4*)(Arow + kk * 64 + half * 16);
      a.q[1] = *(const uint4*)(Arow + kk * 64 + 32 + half * 16);
      // All 4 B fragments first, then 4 back-to-back WMMAs: a single DS wait
      // covers the group and the XDLs can pipeline.
      BF16x16 b[4];
      for (int t = 0; t < 4; ++t) {
        const char* Brow = Bt + (t * 16 + l16) * TILE_ROW_B + kk * 64 + half * 16;
        b[t].q[0] = ((const uint4*)Brow)[0];
        b[t].q[1] = ((const uint4*)Brow)[2];
      }
      for (int t = 0; t < 4; ++t)
        acc[t] = __builtin_amdgcn_wmma_f32_16x16x32_bf16(
                   false, a.v, false, b[t].v, (short)0, acc[t], false, false);
    }

    // ---- publish stage s+1, retire buffer s ----
    if (s + 1 < nstages) {
#if USE_TDM
      if (tid < 32) __builtin_amdgcn_s_wait_tensorcnt(0);
#endif
      __syncthreads();
    }
  }

  // C/D layout: VGPR j <-> row m0+16*wave + j + 8*half, column n0+16t+l16.
  for (int t = 0; t < 4; ++t) {
    const int n  = n0 + t * 16 + l16;
    const float bv = bias ? bias[n] : 0.f;
    for (int j = 0; j < 8; ++j) {
      const int m = m0 + wave * 16 + j + half * 8;
      float v = acc[t][j] + bv;
      if (act == 1) v = tanhf(v);
      C[(size_t)m * N + n] = v;
      if (Cbf) Cbf[(size_t)m * N + n] = f2bf(v);
    }
  }
}

// ---------------------------------------------------------------------------
// Elementwise / gather helpers (grid-stride)
// ---------------------------------------------------------------------------
__global__ void k_cvt(const float* __restrict__ src, u16* __restrict__ dst,
                      long rows, long cols, long ld)
{
  long total = rows * cols;
  for (long i = (long)blockIdx.x * blockDim.x + threadIdx.x; i < total;
       i += (long)gridDim.x * blockDim.x) {
    long r = i / cols, c = i - r * cols;
    dst[i] = f2bf(src[r * ld + c]);
  }
}

__global__ void k_gather(const int* __restrict__ tok, const float* __restrict__ tab,
                         u16* __restrict__ out, int Srows, int Bn, int En, int rev)
{
  long total = (long)Srows * Bn * En;
  for (long i = (long)blockIdx.x * blockDim.x + threadIdx.x; i < total;
       i += (long)gridDim.x * blockDim.x) {
    int  e = (int)(i % En);
    long t = i / En;
    int  b = (int)(t % Bn);
    int  s = (int)(t / Bn);
    int ss = rev ? (Srows - 1 - s) : s;
    int id = tok[ss * Bn + b];
    out[i] = f2bf(tab[(long)id * En + e]);
  }
}

// GRU gate fusion: h2 = (1-z)*n + z*h ; writes fp32 state + bf16 mirror,
// optionally the encoder output slot and the deferred decoder feature row.
__global__ void k_gru(const float* __restrict__ gx, const float* __restrict__ gh,
                      float* __restrict__ h, u16* __restrict__ hbf,
                      float* __restrict__ enc, u16* __restrict__ encbf, long enc_stride,
                      u16* __restrict__ feat, int Bn, int Hn)
{
  long total = (long)Bn * Hn;
  for (long i = (long)blockIdx.x * blockDim.x + threadIdx.x; i < total;
       i += (long)gridDim.x * blockDim.x) {
    int b = (int)(i / Hn), j = (int)(i % Hn);
    long g = (long)b * 3 * Hn + j;
    float xr = gx[g], xz = gx[g + Hn], xn = gx[g + 2 * Hn];
    float hr = gh[g], hz = gh[g + Hn], hn = gh[g + 2 * Hn];
    float hp = h[i];
    float r  = sigmoidf_(xr + hr);
    float z  = sigmoidf_(xz + hz);
    float n  = tanhf(xn + r * hn);
    float h2 = (1.f - z) * n + z * hp;
    h[i]   = h2;
    hbf[i] = f2bf(h2);
    if (enc)  { enc[(long)b * enc_stride + j]   = h2;
                encbf[(long)b * enc_stride + j] = f2bf(h2); }
    if (feat) feat[(long)b * 3584 + j] = f2bf(h2);   // [h2 | w | et] row
  }
}

// Masked attention scores + softmax. One block per batch element b.
// sc[s] = v . tanh(proj[b] + Ue[b,s])   (b_attn folded into Ue's bias)
__global__ __launch_bounds__(256)
void k_attn_score(const float* __restrict__ proj, const float* __restrict__ Ue,
                  const float* __restrict__ vattn, const int* __restrict__ src,
                  float* __restrict__ a_out)
{
  const int b = blockIdx.x;
  __shared__ float sc[48];
  const int wave = threadIdx.x >> 5, lane = threadIdx.x & 31;
  const float* pr = proj + (long)b * 1024;
  for (int s = wave; s < 48; s += 8) {
    const float* ue = Ue + ((long)b * 48 + s) * 1024;
    float p = 0.f;
    for (int j = lane; j < 1024; j += 32)
      p += vattn[j] * tanhf(pr[j] + ue[j]);
    for (int o = 16; o; o >>= 1) p += __shfl_down(p, o, 32);
    if (lane == 0) sc[s] = p;
  }
  __syncthreads();
  if (threadIdx.x < 32) {
    int lane2 = threadIdx.x;
    int s0 = lane2, s1 = lane2 + 32;
    float m0 = (src[s0 * 64 + b] != 1) ? sc[s0] : -1e10f;
    float m1 = (s1 < 48) ? ((src[s1 * 64 + b] != 1) ? sc[s1] : -1e10f) : -3.4e38f;
    float mx = fmaxf(m0, m1);
    for (int o = 16; o; o >>= 1) mx = fmaxf(mx, __shfl_xor(mx, o, 32));
    float e0 = __expf(m0 - mx);
    float e1 = (s1 < 48) ? __expf(m1 - mx) : 0.f;
    float sm = e0 + e1;
    for (int o = 16; o; o >>= 1) sm += __shfl_xor(sm, o, 32);
    a_out[b * 48 + s0] = e0 / sm;
    if (s1 < 48) a_out[b * 48 + s1] = e1 / sm;
  }
}

// Context w[b,:] = sum_s a[b,s] * enc_bt[b,s,:]; write bf16 into decoder
// GEMM input x_d[b][512:2560] and deferred feature row feat[b][1024:3072].
__global__ __launch_bounds__(256)
void k_context(const float* __restrict__ a, const float* __restrict__ enc,
               u16* __restrict__ x_d, u16* __restrict__ feat)
{
  const int b = blockIdx.x;
  for (int h = threadIdx.x; h < 2048; h += 256) {
    float acc = 0.f;
    const float* eb = enc + (long)b * 48 * 2048 + h;
    for (int s = 0; s < 48; ++s) acc += a[b * 48 + s] * eb[(long)s * 2048];
    u16 u = f2bf(acc);
    x_d[(long)b * 2560 + 512 + h]  = u;
    feat[(long)b * 3584 + 1024 + h] = u;
  }
}

// Place decoder-token embedding into x_d[b][0:512] and feat[b][3072:3584].
__global__ void k_place_et(const u16* __restrict__ et, u16* __restrict__ x_d,
                           u16* __restrict__ feat)
{
  long total = 64L * 512;
  for (long i = (long)blockIdx.x * blockDim.x + threadIdx.x; i < total;
       i += (long)gridDim.x * blockDim.x) {
    int b = (int)(i >> 9), j = (int)(i & 511);
    u16 u = et[i];
    x_d[(long)b * 2560 + j]         = u;
    feat[(long)b * 3584 + 3072 + j] = u;
  }
}

// hcat = bf16([hf | hb])  (64 x 2048)
__global__ void k_hcat(const float* __restrict__ hf, const float* __restrict__ hb,
                       u16* __restrict__ out)
{
  long total = 64L * 2048;
  for (long i = (long)blockIdx.x * blockDim.x + threadIdx.x; i < total;
       i += (long)gridDim.x * blockDim.x) {
    int b = (int)(i / 2048), j = (int)(i % 2048);
    float v = (j < 1024) ? hf[b * 1024 + j] : hb[b * 1024 + j - 1024];
    out[i] = f2bf(v);
  }
}

// ---------------------------------------------------------------------------
extern "C" void kernel_launch(void* const* d_in, const int* in_sizes, int n_in,
                              void* d_out, int out_size, void* d_ws, size_t ws_size,
                              hipStream_t stream)
{
  (void)in_sizes; (void)n_in; (void)out_size; (void)ws_size;

  const int   S = 48, B = 64, H = 1024, E = 512, V = 32000, TD = 31;
  const int   G = 3 * H;             // 3072
  const int   KD = 2 * H + E;        // 2560
  const int   KO = 3 * H + E;        // 3584

  const int*   src     = (const int*)  d_in[0];
  const int*   trg     = (const int*)  d_in[1];
  const float* emb_enc = (const float*)d_in[2];
  const float* Wi_f    = (const float*)d_in[3];
  const float* Wh_f    = (const float*)d_in[4];
  const float* bi_f    = (const float*)d_in[5];
  const float* bh_f    = (const float*)d_in[6];
  const float* Wi_b    = (const float*)d_in[7];
  const float* Wh_b    = (const float*)d_in[8];
  const float* bi_b    = (const float*)d_in[9];
  const float* bh_b    = (const float*)d_in[10];
  const float* W_fc    = (const float*)d_in[11];
  const float* b_fc    = (const float*)d_in[12];
  const float* W_attn  = (const float*)d_in[13];
  const float* b_attn  = (const float*)d_in[14];
  const float* v_attn  = (const float*)d_in[15];
  const float* emb_dec = (const float*)d_in[16];
  const float* Wi_d    = (const float*)d_in[17];
  const float* Wh_d    = (const float*)d_in[18];
  const float* bi_d    = (const float*)d_in[19];
  const float* bh_d    = (const float*)d_in[20];
  const float* W_out   = (const float*)d_in[21];
  const float* b_out   = (const float*)d_in[22];

  // ---- workspace carving (256B aligned) ----
  char*  base = (char*)d_ws;
  size_t off  = 0;
  auto alloc = [&](size_t bytes) -> void* {
    void* p = base + off;
    off = (off + bytes + 255) & ~(size_t)255;
    return p;
  };
  u16* wWi_f  = (u16*)alloc((size_t)G * E  * 2);
  u16* wWh_f  = (u16*)alloc((size_t)G * H  * 2);
  u16* wWi_b  = (u16*)alloc((size_t)G * E  * 2);
  u16* wWh_b  = (u16*)alloc((size_t)G * H  * 2);
  u16* wW_fc  = (u16*)alloc((size_t)H * 2*H * 2);
  u16* wWa_h  = (u16*)alloc((size_t)H * H   * 2);
  u16* wWa_e  = (u16*)alloc((size_t)H * 2*H * 2);
  u16* wWi_d  = (u16*)alloc((size_t)G * KD * 2);
  u16* wWh_d  = (u16*)alloc((size_t)G * H  * 2);
  u16* wW_out = (u16*)alloc((size_t)V * KO * 2);

  u16*   emb_f   = (u16*)  alloc((size_t)S * B * E * 2);
  u16*   emb_r   = (u16*)  alloc((size_t)S * B * E * 2);
  u16*   et_all  = (u16*)  alloc((size_t)TD * B * E * 2);
  float* gx_f    = (float*)alloc((size_t)S * B * G * 4);
  float* gx_b    = (float*)alloc((size_t)S * B * G * 4);
  float* gh      = (float*)alloc((size_t)B * G * 4);
  float* h_f     = (float*)alloc((size_t)B * H * 4);
  float* h_b     = (float*)alloc((size_t)B * H * 4);
  u16*   hf_bf   = (u16*)  alloc((size_t)B * H * 2);
  u16*   hb_bf   = (u16*)  alloc((size_t)B * H * 2);
  float* enc_bt  = (float*)alloc((size_t)B * S * 2*H * 4);
  u16*   enc_bf  = (u16*)  alloc((size_t)B * S * 2*H * 2);
  u16*   hcat    = (u16*)  alloc((size_t)B * 2*H * 2);
  float* h_dec   = (float*)alloc((size_t)B * H * 4);
  u16*   h_dec_bf= (u16*)  alloc((size_t)B * H * 2);
  float* Ue      = (float*)alloc((size_t)B * S * H * 4);
  float* proj    = (float*)alloc((size_t)B * H * 4);
  float* a_w     = (float*)alloc((size_t)B * S * 4);
  u16*   x_d     = (u16*)  alloc((size_t)B * KD * 2);
  float* gx_d    = (float*)alloc((size_t)B * G * 4);
  u16*   dec_feat= (u16*)  alloc((size_t)TD * B * KO * 2);

  auto blocks = [](long n) -> int {
    long b = (n + 255) / 256;
    return (int)(b > 65535 ? 65535 : b);
  };
  auto gemm = [&](const u16* A, const u16* W, const float* bias, float* C,
                  u16* Cbf, int M, int N, int K, int act) {
    dim3 g(N / 64, M / 64);
    k_gemm<<<g, 128, 0, stream>>>(A, W, bias, C, Cbf, M, N, K, act);
  };

  // ---- zero init (capture-safe memsets) ----
  hipMemsetAsync(d_out, 0, (size_t)B * V * 4, stream);       // outputs[0] = 0
  hipMemsetAsync(h_f,   0, (size_t)B * H * 4, stream);
  hipMemsetAsync(h_b,   0, (size_t)B * H * 4, stream);
  hipMemsetAsync(hf_bf, 0, (size_t)B * H * 2, stream);
  hipMemsetAsync(hb_bf, 0, (size_t)B * H * 2, stream);

  // ---- weight conversion to bf16 (W_attn split into [W_h | W_enc]) ----
  k_cvt<<<blocks((long)G*E),  256, 0, stream>>>(Wi_f,  wWi_f,  G, E,  E);
  k_cvt<<<blocks((long)G*H),  256, 0, stream>>>(Wh_f,  wWh_f,  G, H,  H);
  k_cvt<<<blocks((long)G*E),  256, 0, stream>>>(Wi_b,  wWi_b,  G, E,  E);
  k_cvt<<<blocks((long)G*H),  256, 0, stream>>>(Wh_b,  wWh_b,  G, H,  H);
  k_cvt<<<blocks((long)H*2*H),256, 0, stream>>>(W_fc,  wW_fc,  H, 2*H, 2*H);
  k_cvt<<<blocks((long)H*H),  256, 0, stream>>>(W_attn,        wWa_h, H, H,   G);
  k_cvt<<<blocks((long)H*2*H),256, 0, stream>>>(W_attn + H,    wWa_e, H, 2*H, G);
  k_cvt<<<blocks((long)G*KD), 256, 0, stream>>>(Wi_d,  wWi_d,  G, KD, KD);
  k_cvt<<<blocks((long)G*H),  256, 0, stream>>>(Wh_d,  wWh_d,  G, H,  H);
  k_cvt<<<blocks((long)V*KO), 256, 0, stream>>>(W_out, wW_out, V, KO, KO);

  // ---- embedding gathers (fwd, reversed, decoder trg[:-1]) ----
  k_gather<<<blocks((long)S*B*E),  256, 0, stream>>>(src, emb_enc, emb_f, S, B, E, 0);
  k_gather<<<blocks((long)S*B*E),  256, 0, stream>>>(src, emb_enc, emb_r, S, B, E, 1);
  k_gather<<<blocks((long)TD*B*E), 256, 0, stream>>>(trg, emb_dec, et_all, TD, B, E, 0);

  // ---- encoder input projections (big WMMA GEMMs, 3072x512x3072) ----
  gemm(emb_f, wWi_f, bi_f, gx_f, nullptr, S * B, G, E, 0);
  gemm(emb_r, wWi_b, bi_b, gx_b, nullptr, S * B, G, E, 0);

  // ---- bi-GRU encoder scan ----
  for (int s = 0; s < S; ++s) {
    gemm(hf_bf, wWh_f, bh_f, gh, nullptr, B, G, H, 0);
    k_gru<<<256, 256, 0, stream>>>(gx_f + (size_t)s * B * G, gh, h_f, hf_bf,
                                   enc_bt + (size_t)s * 2 * H,
                                   enc_bf + (size_t)s * 2 * H,
                                   (long)S * 2 * H, nullptr, B, H);
    gemm(hb_bf, wWh_b, bh_b, gh, nullptr, B, G, H, 0);
    k_gru<<<256, 256, 0, stream>>>(gx_b + (size_t)s * B * G, gh, h_b, hb_bf,
                                   enc_bt + (size_t)(S - 1 - s) * 2 * H + H,
                                   enc_bf + (size_t)(S - 1 - s) * 2 * H + H,
                                   (long)S * 2 * H, nullptr, B, H);
  }

  // ---- decoder init hidden = tanh([hf|hb] @ W_fc^T + b_fc) ----
  k_hcat<<<blocks(64L * 2048), 256, 0, stream>>>(h_f, h_b, hcat);
  gemm(hcat, wW_fc, b_fc, h_dec, h_dec_bf, B, H, 2 * H, 1);

  // ---- hoisted attention encoder projection: Ue = enc @ W_enc^T + b_attn ----
  gemm(enc_bf, wWa_e, b_attn, Ue, nullptr, B * S, H, 2 * H, 0);

  // ---- decoder scan (31 steps); vocab projection deferred ----
  for (int t = 0; t < TD; ++t) {
    u16* feat = dec_feat + (size_t)t * B * KO;
    gemm(h_dec_bf, wWa_h, nullptr, proj, nullptr, B, H, H, 0);
    k_attn_score<<<B, 256, 0, stream>>>(proj, Ue, v_attn, src, a_w);
    k_context<<<B, 256, 0, stream>>>(a_w, enc_bt, x_d, feat);
    k_place_et<<<blocks(64L * 512), 256, 0, stream>>>(et_all + (size_t)t * B * E, x_d, feat);
    gemm(x_d, wWi_d, bi_d, gx_d, nullptr, B, G, KD, 0);
    gemm(h_dec_bf, wWh_d, bh_d, gh, nullptr, B, G, H, 0);
    k_gru<<<256, 256, 0, stream>>>(gx_d, gh, h_dec, h_dec_bf,
                                   nullptr, nullptr, 0, feat, B, H);
  }

  // ---- one batched vocab projection: (31*64) x 3584 x 32000 ----
  gemm(dec_feat, wW_out, b_out, (float*)d_out + (size_t)B * V, nullptr,
       TD * B, V, KO, 0);
}
